// ISDALoss_71330816852541
// MI455X (gfx1250) — compile-verified, matrix operand order checked by reading, not run on hardware
//
#include <hip/hip_runtime.h>
#include <hip/hip_bf16.h>
#include <cstdint>

// Problem sizes (fixed by the reference)
#define NN 4096
#define AA 512
#define CC 8192

typedef __attribute__((ext_vector_type(16))) __bf16 v16bf;
typedef __attribute__((ext_vector_type(8)))  float  v8f;
typedef __attribute__((ext_vector_type(4)))  unsigned int u32x4;
typedef __attribute__((ext_vector_type(4)))  int          i32x4;
typedef __attribute__((ext_vector_type(8)))  int          i32x8;

union FragBF { uint4 q[2]; v16bf v; };

#if defined(__has_builtin)
#if __has_builtin(__builtin_amdgcn_tensor_load_to_lds)
#define HAVE_TDM 1
#endif
#endif
#ifndef HAVE_TDM
#define HAVE_TDM 0
#endif

// float -> bf16 bits, round-to-nearest-even
__device__ __forceinline__ uint16_t bfbits(float f) {
  union { float f; uint32_t u; } x; x.f = f;
  return (uint16_t)((x.u + 0x7FFFu + ((x.u >> 16) & 1u)) >> 16);
}
__device__ __forceinline__ uint2 pack4(float4 v) {
  uint2 r;
  r.x = (uint32_t)bfbits(v.x) | ((uint32_t)bfbits(v.y) << 16);
  r.y = (uint32_t)bfbits(v.z) | ((uint32_t)bfbits(v.w) << 16);
  return r;
}

#if HAVE_TDM
// Issue a 2D TDM tile load: tile_h rows x tile_w bf16 elements from a row-major
// bf16 tensor (row stride in elements) into LDS (packed [row][tile_w]).
// D# bitfields per CDNA5 ISA.
__device__ __forceinline__ void tdm_load_2d(const uint16_t* gsrc, uint32_t lds_off,
                                            uint32_t tile_w, uint32_t tile_h,
                                            uint64_t row_stride_elems,
                                            uint32_t tensor_w, uint32_t tensor_h) {
  uint64_t ga = (uint64_t)(uintptr_t)gsrc;
  u32x4 g0;
  g0[0] = 1u;                                            // count=1, user descriptor
  g0[1] = lds_off;                                       // lds_addr (bytes)
  g0[2] = (uint32_t)(ga & 0xFFFFFFFFull);                // global_addr[31:0]
  g0[3] = (uint32_t)((ga >> 32) & 0x01FFFFFFull) | 0x80000000u; // [56:32] | type=2
  i32x8 g1;
  g1[0] = (int)0x00010000u;                              // data_size=1 (2 bytes)
  g1[1] = (int)((tensor_w & 0xFFFFu) << 16);             // tensor_dim0[15:0] @ [63:48]
  g1[2] = (int)((tensor_w >> 16) | ((tensor_h & 0xFFFFu) << 16)); // dim0 hi | dim1 lo
  g1[3] = (int)((tensor_h >> 16) | (tile_w << 16));      // dim1 hi | tile_dim0
  g1[4] = (int)tile_h;                                   // tile_dim1 (tile_dim2=0)
  g1[5] = (int)(uint32_t)(row_stride_elems & 0xFFFFFFFFull); // dim0_stride[31:0]
  g1[6] = (int)(uint32_t)((row_stride_elems >> 32) & 0xFFFFull); // dim0_stride[47:32]
  g1[7] = 0;
  i32x4 z4 = {0, 0, 0, 0};
#if __clang_major__ >= 23
  i32x8 z8 = {0, 0, 0, 0, 0, 0, 0, 0};
  __builtin_amdgcn_tensor_load_to_lds(g0, g1, z4, z4, z8, 0);
#else
  __builtin_amdgcn_tensor_load_to_lds(g0, g1, z4, z4, 0);
#endif
}
#endif

// Cooperative fallback staging: copy tile_h x 64 bf16 (k-contiguous) to LDS.
__device__ __forceinline__ void stage_tile64(uint16_t* dst, const uint16_t* src,
                                             int tile_h, size_t row_stride, int t) {
  int n8 = tile_h * 8;  // 8 bf16 per 16-byte chunk, 8 chunks per 64-wide row
  for (int e = t; e < n8; e += 256) {
    int row = e >> 3, part = e & 7;
    *reinterpret_cast<uint4*>(dst + row * 64 + part * 8) =
        *reinterpret_cast<const uint4*>(src + (size_t)row * row_stride + part * 8);
  }
}

// ---------------------------------------------------------------- init / CV
__global__ void k_init(float* __restrict__ wsz, int nzero, int* __restrict__ winner) {
  int idx = blockIdx.x * blockDim.x + threadIdx.x;
  if (idx < nzero) wsz[idx] = 0.0f;
  if (idx < CC) winner[idx] = -1;
}

__global__ void k_count(const int* __restrict__ labels, const int* __restrict__ index_tail,
                        float* __restrict__ cnt, int* __restrict__ winner) {
  int i = blockIdx.x * blockDim.x + threadIdx.x;
  if (i >= NN) return;
  atomicAdd(&cnt[labels[i]], 1.0f);
  atomicMax(&winner[index_tail[i]], i);   // deterministic "last occurrence wins"
}

__global__ void k_sum(const float* __restrict__ feat, const int* __restrict__ labels,
                      float* __restrict__ sum) {
  int e4 = blockIdx.x * blockDim.x + threadIdx.x;
  if (e4 >= NN * AA / 4) return;
  int e = e4 * 4;
  int i = e >> 9, a = e & (AA - 1);
  int lab = labels[i];
  float4 v = *reinterpret_cast<const float4*>(&feat[e]);
  float* dst = &sum[(size_t)lab * AA + a];
  atomicAdd(dst + 0, v.x); atomicAdd(dst + 1, v.y);
  atomicAdd(dst + 2, v.z); atomicAdd(dst + 3, v.w);
}

__global__ void k_var(const float* __restrict__ feat, const int* __restrict__ labels,
                      const float* __restrict__ cnt, const float* __restrict__ sum,
                      float* __restrict__ varsum) {
  int e4 = blockIdx.x * blockDim.x + threadIdx.x;
  if (e4 >= NN * AA / 4) return;
  int e = e4 * 4;
  int i = e >> 9, a = e & (AA - 1);
  int lab = labels[i];
  float cs = fmaxf(cnt[lab], 1.0f);
  float4 f = *reinterpret_cast<const float4*>(&feat[e]);
  float4 s = *reinterpret_cast<const float4*>(&sum[(size_t)lab * AA + a]);
  float* dst = &varsum[(size_t)lab * AA + a];
  float d0 = f.x - s.x / cs, d1 = f.y - s.y / cs, d2 = f.z - s.z / cs, d3 = f.w - s.w / cs;
  atomicAdd(dst + 0, d0 * d0); atomicAdd(dst + 1, d1 * d1);
  atomicAdd(dst + 2, d2 * d2); atomicAdd(dst + 3, d3 * d3);
}

__global__ void k_finalize(const float* __restrict__ cnt, const float* __restrict__ sum,
                           const float* __restrict__ varsum, const float* __restrict__ cov,
                           const float* __restrict__ ave, const float* __restrict__ amount,
                           float* __restrict__ ncov_ws, float* __restrict__ out_cv,
                           float* __restrict__ out_ave, float* __restrict__ out_amt) {
  int idx = blockIdx.x * blockDim.x + threadIdx.x;
  if (idx >= CC * AA) return;
  int c = idx >> 9;
  float cv = cnt[c];
  float cs = fmaxf(cv, 1.0f);
  float av = sum[idx] / cs;
  float vr = varsum[idx] / cs;
  float denom = cv + amount[c];
  float w = (denom > 0.0f) ? (cv / fmaxf(denom, 1.0f)) : 0.0f;
  float d = ave[idx] - av;
  float nc = cov[idx] * (1.0f - w) + vr * w + w * (1.0f - w) * d * d;
  float na = ave[idx] * (1.0f - w) + av * w;
  ncov_ws[idx] = nc;   // aligned mirror of cv_var (tail rows patched later)
  out_cv[idx]  = nc;
  out_ave[idx] = na;
  if ((idx & (AA - 1)) == 0) out_amt[c] = amount[c] + cv;
}

// ------------------------------------------------------- bf16 conversions
__global__ void k_conv_kg(const float* __restrict__ kg, const int* __restrict__ index_tail,
                          uint16_t* __restrict__ kgb) {
  size_t e4 = (size_t)blockIdx.x * blockDim.x + threadIdx.x;
  if (e4 >= (size_t)NN * CC / 4) return;
  size_t e = e4 * 4;
  int i = (int)(e >> 13);
  int c = (int)(e & (CC - 1));
  int row = index_tail[i];
  float4 v = *reinterpret_cast<const float4*>(&kg[(size_t)row * CC + c]);
  *reinterpret_cast<uint2*>(&kgb[e]) = pack4(v);
}

__global__ void k_conv_W(const float* __restrict__ fc, uint16_t* __restrict__ W2b,
                         uint16_t* __restrict__ Wb) {
  size_t e4 = (size_t)blockIdx.x * blockDim.x + threadIdx.x;
  if (e4 >= (size_t)CC * AA / 4) return;
  size_t e = e4 * 4;
  float4 v = *reinterpret_cast<const float4*>(&fc[e]);
  float4 v2 = {v.x * v.x, v.y * v.y, v.z * v.z, v.w * v.w};
  *reinterpret_cast<uint2*>(&Wb[e])  = pack4(v);
  *reinterpret_cast<uint2*>(&W2b[e]) = pack4(v2);
}

// ncovT[a*CC + c] = bf16(ncov[c*AA + a]) via 32x32 LDS tile transpose
__global__ __launch_bounds__(256)
void k_conv_ncovT(const float* __restrict__ ncov, uint16_t* __restrict__ ncovT) {
  __shared__ uint16_t s[32][33];
  int c0 = blockIdx.x * 32, a0 = blockIdx.y * 32;
  int t = threadIdx.x;
  for (int e = t; e < 1024; e += 256) {
    int cl = e >> 5, al = e & 31;
    s[cl][al] = bfbits(ncov[(size_t)(c0 + cl) * AA + a0 + al]);
  }
  __syncthreads();
  for (int e = t; e < 1024; e += 256) {
    int al = e >> 5, cl = e & 31;
    ncovT[(size_t)(a0 + al) * CC + c0 + cl] = s[cl][al];
  }
}

// A1b = bf16(cv_var[labels]); A2b = bf16(cv_var[labels]*W[labels])
__global__ void k_conv_augA(const float* __restrict__ cvv, const float* __restrict__ fc,
                            const int* __restrict__ labels, uint16_t* __restrict__ A1b,
                            uint16_t* __restrict__ A2b) {
  int e4 = blockIdx.x * blockDim.x + threadIdx.x;
  if (e4 >= NN * AA / 4) return;
  int e = e4 * 4;
  int i = e >> 9, a = e & (AA - 1);
  int lab = labels[i];
  float4 c = *reinterpret_cast<const float4*>(&cvv[(size_t)lab * AA + a]);
  float4 w = *reinterpret_cast<const float4*>(&fc[(size_t)lab * AA + a]);
  float4 cw = {c.x * w.x, c.y * w.y, c.z * w.z, c.w * w.w};
  *reinterpret_cast<uint2*>(&A1b[e]) = pack4(c);
  *reinterpret_cast<uint2*>(&A2b[e]) = pack4(cw);
}

// ------------------------------------------- GEMM 1: cv_tail + tail scatter
// D(4096x512) = kgb (4096x8192 bf16) x ncov (8192x512, staged as ncovT).
// 64(M) x 32(N) block, BK=64 (2 WMMA K-steps per staged buffer), 8 waves,
// one 16x16 tile/wave, TDM double buffer.
__global__ __launch_bounds__(256)
void k_gemm_tail(const uint16_t* __restrict__ kgb, const uint16_t* __restrict__ ncovT,
                 const int* __restrict__ index_tail, const int* __restrict__ winner,
                 float* __restrict__ out_cv, float* __restrict__ cvv_ws) {
  __shared__ __align__(16) uint16_t sA[2][64 * 64];
  __shared__ __align__(16) uint16_t sB[2][32 * 64];
  int t = threadIdx.x;
  int bm = blockIdx.y, bn = blockIdx.x;
  int w = t >> 5, l = t & 31;
  int tm = w & 3, tn = w >> 2;
  int lr = l & 15, lh = l >> 4, kb = lh * 8;
  const uint16_t* gA = kgb + (size_t)bm * 64 * CC;
  const uint16_t* gB = ncovT + (size_t)bn * 32 * CC;
  v8f acc = {};
  const int iters = CC / 64;

#if HAVE_TDM
  bool issuer = (w == 0);
  uint32_t ldsA[2] = {(uint32_t)(uintptr_t)&sA[0][0], (uint32_t)(uintptr_t)&sA[1][0]};
  uint32_t ldsB[2] = {(uint32_t)(uintptr_t)&sB[0][0], (uint32_t)(uintptr_t)&sB[1][0]};
  if (issuer) {
    tdm_load_2d(gA, ldsA[0], 64, 64, CC, CC, NN);
    tdm_load_2d(gB, ldsB[0], 64, 32, CC, CC, AA);
    __builtin_amdgcn_s_wait_tensorcnt(0);
  }
  __syncthreads();
  for (int it = 0; it < iters; ++it) {
    int buf = it & 1;
    if (issuer && it + 1 < iters) {
      int kk = (it + 1) * 64;
      tdm_load_2d(gA + kk, ldsA[buf ^ 1], 64, 64, CC, CC, NN);
      tdm_load_2d(gB + kk, ldsB[buf ^ 1], 64, 32, CC, CC, AA);
    }
    const uint16_t* pa = &sA[buf][(tm * 16 + lr) * 64];
    const uint16_t* pb = &sB[buf][(tn * 16 + lr) * 64];
#pragma unroll
    for (int ks = 0; ks < 2; ++ks) {
      FragBF fa, fb;
      fa.q[0] = *reinterpret_cast<const uint4*>(pa + ks * 32 + kb);
      fa.q[1] = *reinterpret_cast<const uint4*>(pa + ks * 32 + kb + 16);
      fb.q[0] = *reinterpret_cast<const uint4*>(pb + ks * 32 + kb);
      fb.q[1] = *reinterpret_cast<const uint4*>(pb + ks * 32 + kb + 16);
      acc = __builtin_amdgcn_wmma_f32_16x16x32_bf16(false, fa.v, false, fb.v,
                                                    (short)0, acc, false, false);
    }
    if (issuer && it + 1 < iters) __builtin_amdgcn_s_wait_tensorcnt(0);
    __syncthreads();
  }
#else
  for (int it = 0; it < iters; ++it) {
    int kk = it * 64;
    stage_tile64(&sA[0][0], gA + kk, 64, CC, t);
    stage_tile64(&sB[0][0], gB + kk, 32, CC, t);
    __syncthreads();
    const uint16_t* pa = &sA[0][(tm * 16 + lr) * 64];
    const uint16_t* pb = &sB[0][(tn * 16 + lr) * 64];
#pragma unroll
    for (int ks = 0; ks < 2; ++ks) {
      FragBF fa, fb;
      fa.q[0] = *reinterpret_cast<const uint4*>(pa + ks * 32 + kb);
      fa.q[1] = *reinterpret_cast<const uint4*>(pa + ks * 32 + kb + 16);
      fb.q[0] = *reinterpret_cast<const uint4*>(pb + ks * 32 + kb);
      fb.q[1] = *reinterpret_cast<const uint4*>(pb + ks * 32 + kb + 16);
      acc = __builtin_amdgcn_wmma_f32_16x16x32_bf16(false, fa.v, false, fb.v,
                                                    (short)0, acc, false, false);
    }
    __syncthreads();
  }
#endif
  for (int r = 0; r < 8; ++r) {
    int m = r + lh * 8;
    int i = bm * 64 + tm * 16 + m;
    int grow = index_tail[i];
    if (winner[grow] == i) {
      int a = bn * 32 + tn * 16 + lr;
      out_cv[(size_t)grow * AA + a]  = acc[r];
      cvv_ws[(size_t)grow * AA + a] = acc[r];   // aligned mirror
    }
  }
}

// ---------------------------------------------------------------- term3
__global__ __launch_bounds__(256)
void k_term3(const float* __restrict__ cvv, const float* __restrict__ fc,
             const int* __restrict__ labels, float* __restrict__ term3) {
  int i = blockIdx.x, t = threadIdx.x;
  int lab = labels[i];
  float s = 0.0f;
  for (int a = t; a < AA; a += 256) {
    float c  = cvv[(size_t)lab * AA + a];
    float wv = fc[(size_t)lab * AA + a];
    s += c * wv * wv;
  }
  __shared__ float red[256];
  red[t] = s; __syncthreads();
  for (int o = 128; o > 0; o >>= 1) { if (t < o) red[t] += red[t + o]; __syncthreads(); }
  if (t == 0) term3[i] = red[0];
}

// -------------------------------- GEMM 2: aug_y = y + 0.5*beta*(t1-2*t2+t3)
// BK=64: 4 WMMA (2 accumulators x 2 K-steps) per staged buffer; 8 K-iters.
__global__ __launch_bounds__(256)
void k_gemm_aug(const uint16_t* __restrict__ A1b, const uint16_t* __restrict__ A2b,
                const uint16_t* __restrict__ W2b, const uint16_t* __restrict__ Wb,
                const float* __restrict__ y, const float* __restrict__ term3,
                const float* __restrict__ beta, float* __restrict__ out_aug) {
  __shared__ __align__(16) uint16_t sA1[2][64 * 64];
  __shared__ __align__(16) uint16_t sA2[2][64 * 64];
  __shared__ __align__(16) uint16_t sB1[2][32 * 64];
  __shared__ __align__(16) uint16_t sB2[2][32 * 64];
  int t = threadIdx.x;
  int bm = blockIdx.y, bn = blockIdx.x;
  int w = t >> 5, l = t & 31;
  int tm = w & 3, tn = w >> 2;
  int lr = l & 15, lh = l >> 4, kb = lh * 8;
  const uint16_t* gA1 = A1b + (size_t)bm * 64 * AA;
  const uint16_t* gA2 = A2b + (size_t)bm * 64 * AA;
  const uint16_t* gB1 = W2b + (size_t)bn * 32 * AA;
  const uint16_t* gB2 = Wb  + (size_t)bn * 32 * AA;
  v8f acc1 = {}, acc2 = {};
  const int iters = AA / 64;

#if HAVE_TDM
  bool issuer = (w == 0);
  uint32_t lA1[2] = {(uint32_t)(uintptr_t)&sA1[0][0], (uint32_t)(uintptr_t)&sA1[1][0]};
  uint32_t lA2[2] = {(uint32_t)(uintptr_t)&sA2[0][0], (uint32_t)(uintptr_t)&sA2[1][0]};
  uint32_t lB1[2] = {(uint32_t)(uintptr_t)&sB1[0][0], (uint32_t)(uintptr_t)&sB1[1][0]};
  uint32_t lB2[2] = {(uint32_t)(uintptr_t)&sB2[0][0], (uint32_t)(uintptr_t)&sB2[1][0]};
  if (issuer) {
    tdm_load_2d(gA1, lA1[0], 64, 64, AA, AA, NN);
    tdm_load_2d(gA2, lA2[0], 64, 64, AA, AA, NN);
    tdm_load_2d(gB1, lB1[0], 64, 32, AA, AA, CC);
    tdm_load_2d(gB2, lB2[0], 64, 32, AA, AA, CC);
    __builtin_amdgcn_s_wait_tensorcnt(0);
  }
  __syncthreads();
  for (int it = 0; it < iters; ++it) {
    int buf = it & 1;
    if (issuer && it + 1 < iters) {
      int kk = (it + 1) * 64;
      tdm_load_2d(gA1 + kk, lA1[buf ^ 1], 64, 64, AA, AA, NN);
      tdm_load_2d(gA2 + kk, lA2[buf ^ 1], 64, 64, AA, AA, NN);
      tdm_load_2d(gB1 + kk, lB1[buf ^ 1], 64, 32, AA, AA, CC);
      tdm_load_2d(gB2 + kk, lB2[buf ^ 1], 64, 32, AA, AA, CC);
    }
    const uint16_t* pa1 = &sA1[buf][(tm * 16 + lr) * 64];
    const uint16_t* pa2 = &sA2[buf][(tm * 16 + lr) * 64];
    const uint16_t* pb1 = &sB1[buf][(tn * 16 + lr) * 64];
    const uint16_t* pb2 = &sB2[buf][(tn * 16 + lr) * 64];
#pragma unroll
    for (int ks = 0; ks < 2; ++ks) {
      FragBF a1, a2, b1, b2;
      a1.q[0] = *reinterpret_cast<const uint4*>(pa1 + ks * 32 + kb);
      a1.q[1] = *reinterpret_cast<const uint4*>(pa1 + ks * 32 + kb + 16);
      a2.q[0] = *reinterpret_cast<const uint4*>(pa2 + ks * 32 + kb);
      a2.q[1] = *reinterpret_cast<const uint4*>(pa2 + ks * 32 + kb + 16);
      b1.q[0] = *reinterpret_cast<const uint4*>(pb1 + ks * 32 + kb);
      b1.q[1] = *reinterpret_cast<const uint4*>(pb1 + ks * 32 + kb + 16);
      b2.q[0] = *reinterpret_cast<const uint4*>(pb2 + ks * 32 + kb);
      b2.q[1] = *reinterpret_cast<const uint4*>(pb2 + ks * 32 + kb + 16);
      acc1 = __builtin_amdgcn_wmma_f32_16x16x32_bf16(false, a1.v, false, b1.v,
                                                     (short)0, acc1, false, false);
      acc2 = __builtin_amdgcn_wmma_f32_16x16x32_bf16(false, a2.v, false, b2.v,
                                                     (short)0, acc2, false, false);
    }
    if (issuer && it + 1 < iters) __builtin_amdgcn_s_wait_tensorcnt(0);
    __syncthreads();
  }
#else
  for (int it = 0; it < iters; ++it) {
    int kk = it * 64;
    stage_tile64(&sA1[0][0], gA1 + kk, 64, AA, t);
    stage_tile64(&sA2[0][0], gA2 + kk, 64, AA, t);
    stage_tile64(&sB1[0][0], gB1 + kk, 32, AA, t);
    stage_tile64(&sB2[0][0], gB2 + kk, 32, AA, t);
    __syncthreads();
    const uint16_t* pa1 = &sA1[0][(tm * 16 + lr) * 64];
    const uint16_t* pa2 = &sA2[0][(tm * 16 + lr) * 64];
    const uint16_t* pb1 = &sB1[0][(tn * 16 + lr) * 64];
    const uint16_t* pb2 = &sB2[0][(tn * 16 + lr) * 64];
#pragma unroll
    for (int ks = 0; ks < 2; ++ks) {
      FragBF a1, a2, b1, b2;
      a1.q[0] = *reinterpret_cast<const uint4*>(pa1 + ks * 32 + kb);
      a1.q[1] = *reinterpret_cast<const uint4*>(pa1 + ks * 32 + kb + 16);
      a2.q[0] = *reinterpret_cast<const uint4*>(pa2 + ks * 32 + kb);
      a2.q[1] = *reinterpret_cast<const uint4*>(pa2 + ks * 32 + kb + 16);
      b1.q[0] = *reinterpret_cast<const uint4*>(pb1 + ks * 32 + kb);
      b1.q[1] = *reinterpret_cast<const uint4*>(pb1 + ks * 32 + kb + 16);
      b2.q[0] = *reinterpret_cast<const uint4*>(pb2 + ks * 32 + kb);
      b2.q[1] = *reinterpret_cast<const uint4*>(pb2 + ks * 32 + kb + 16);
      acc1 = __builtin_amdgcn_wmma_f32_16x16x32_bf16(false, a1.v, false, b1.v,
                                                     (short)0, acc1, false, false);
      acc2 = __builtin_amdgcn_wmma_f32_16x16x32_bf16(false, a2.v, false, b2.v,
                                                     (short)0, acc2, false, false);
    }
    __syncthreads();
  }
#endif
  float bet = beta[0];
  for (int r = 0; r < 8; ++r) {
    int m = r + lh * 8;
    int i = bm * 64 + tm * 16 + m;
    int k = bn * 32 + tn * 16 + lr;
    float sig = bet * (acc1[r] - 2.0f * acc2[r] + term3[i]);
    size_t idx = (size_t)i * CC + k;
    out_aug[idx] = y[idx] + 0.5f * sig;
  }
}

// --------------------------------------------------------- softmax + loss
__global__ __launch_bounds__(256)
void k_rowloss(const float* __restrict__ aug, const int* __restrict__ labels,
               const float* __restrict__ cw, float* __restrict__ nllw,
               float* __restrict__ wsum) {
  int i = blockIdx.x, t = threadIdx.x;
  const float* row = aug + (size_t)i * CC;
  float m = -3.4e38f;
  for (int j = t; j < CC; j += 256) m = fmaxf(m, row[j]);
  __shared__ float red[256];
  red[t] = m; __syncthreads();
  for (int o = 128; o > 0; o >>= 1) { if (t < o) red[t] = fmaxf(red[t], red[t + o]); __syncthreads(); }
  float rmax = red[0]; __syncthreads();
  float s = 0.0f;
  for (int j = t; j < CC; j += 256) s += expf(row[j] - rmax);
  red[t] = s; __syncthreads();
  for (int o = 128; o > 0; o >>= 1) { if (t < o) red[t] += red[t + o]; __syncthreads(); }
  if (t == 0) {
    int lab = labels[i];
    float lse = rmax + logf(red[0]);
    float wv  = cw[lab];
    nllw[i] = wv * (lse - row[lab]);
    wsum[i] = wv;
  }
}

__global__ __launch_bounds__(256)
void k_loss(const float* __restrict__ nllw, const float* __restrict__ wsum,
            float* __restrict__ out_loss) {
  int t = threadIdx.x;
  float a = 0.0f, b = 0.0f;
  for (int i = t; i < NN; i += 256) { a += nllw[i]; b += wsum[i]; }
  __shared__ float ra[256], rb[256];
  ra[t] = a; rb[t] = b; __syncthreads();
  for (int o = 128; o > 0; o >>= 1) {
    if (t < o) { ra[t] += ra[t + o]; rb[t] += rb[t + o]; }
    __syncthreads();
  }
  if (t == 0) out_loss[0] = ra[0] / rb[0];
}

__global__ void k_copy_y(const float* __restrict__ y, float* __restrict__ out_y) {
  size_t idx = (size_t)blockIdx.x * blockDim.x + threadIdx.x;
  if (idx < (size_t)NN * CC) out_y[idx] = y[idx];
}

// ------------------------------------------------------------------ launch
extern "C" void kernel_launch(void* const* d_in, const int* in_sizes, int n_in,
                              void* d_out, int out_size, void* d_ws, size_t ws_size,
                              hipStream_t stream) {
  (void)in_sizes; (void)n_in; (void)out_size; (void)ws_size;
  const float* features  = (const float*)d_in[0];
  const float* y         = (const float*)d_in[1];
  const int*   labels    = (const int*)d_in[2];
  const float* fc        = (const float*)d_in[3];
  const float* cov       = (const float*)d_in[4];
  const float* ave       = (const float*)d_in[5];
  const float* amount    = (const float*)d_in[6];
  const float* kg_sigma  = (const float*)d_in[7];
  const float* cw        = (const float*)d_in[8];
  const int*   index_tail= (const int*)d_in[9];
  const float* beta      = (const float*)d_in[10];

  // d_out layout: loss(1) | y(N*C) | cv_var(C*A) | new_ave(C*A) | new_amt(C)
  float* out      = (float*)d_out;
  float* out_loss = out;
  float* out_y    = out + 1;
  float* out_cv   = out_y  + (size_t)NN * CC;
  float* out_ave  = out_cv + (size_t)CC * AA;
  float* out_amt  = out_ave + (size_t)CC * AA;

  // workspace layout: f32 region, then int, then bf16 region (all aligned)
  float* ws     = (float*)d_ws;
  float* cnt    = ws;                          // C
  float* sum    = cnt + CC;                    // C*A
  float* varsum = sum + (size_t)CC * AA;       // C*A
  float* ncov   = varsum + (size_t)CC * AA;    // C*A  (becomes aligned cv_var)
  float* term3  = ncov + (size_t)CC * AA;      // N
  float* nllw   = term3 + NN;                  // N
  float* wsum   = nllw + NN;                   // N
  int*   winner = (int*)(wsum + NN);           // C ints
  uint16_t* bfbase = (uint16_t*)(winner + CC);
  uint16_t* kgb   = bfbase;                    // N*C
  uint16_t* ncovT = kgb + (size_t)NN * CC;     // A*C (transposed)
  uint16_t* A1b   = ncovT + (size_t)AA * CC;   // N*A
  uint16_t* A2b   = A1b + (size_t)NN * AA;     // N*A
  uint16_t* W2b   = A2b + (size_t)NN * AA;     // C*A
  uint16_t* Wb    = W2b + (size_t)CC * AA;     // C*A

  int nzero = CC + 2 * CC * AA;                // cnt + sum + varsum
  k_init<<<(nzero + 255) / 256, 256, 0, stream>>>(ws, nzero, winner);
  k_count<<<(NN + 255) / 256, 256, 0, stream>>>(labels, index_tail, cnt, winner);
  k_sum<<<(NN * AA / 4 + 255) / 256, 256, 0, stream>>>(features, labels, sum);
  k_var<<<(NN * AA / 4 + 255) / 256, 256, 0, stream>>>(features, labels, cnt, sum, varsum);
  k_finalize<<<(CC * AA + 255) / 256, 256, 0, stream>>>(cnt, sum, varsum, cov, ave,
                                                        amount, ncov, out_cv, out_ave, out_amt);
  k_conv_W<<<(int)(((size_t)CC * AA / 4 + 255) / 256), 256, 0, stream>>>(fc, W2b, Wb);
  k_conv_kg<<<(int)(((size_t)NN * CC / 4 + 255) / 256), 256, 0, stream>>>(kg_sigma, index_tail, kgb);
  k_conv_ncovT<<<dim3(CC / 32, AA / 32), 256, 0, stream>>>(ncov, ncovT);
  k_gemm_tail<<<dim3(AA / 32, NN / 64), 256, 0, stream>>>(kgb, ncovT, index_tail,
                                                          winner, out_cv, ncov);
  k_conv_augA<<<(NN * AA / 4 + 255) / 256, 256, 0, stream>>>(ncov, fc, labels, A1b, A2b);
  k_term3<<<NN, 256, 0, stream>>>(ncov, fc, labels, term3);
  k_gemm_aug<<<dim3(CC / 32, NN / 64), 256, 0, stream>>>(A1b, A2b, W2b, Wb, y,
                                                         term3, beta, out_y);
  k_rowloss<<<NN, 256, 0, stream>>>(out_y, labels, cw, nllw, wsum);
  k_loss<<<1, 256, 0, stream>>>(nllw, wsum, out_loss);
  k_copy_y<<<(int)(((size_t)NN * CC + 255) / 256), 256, 0, stream>>>(y, out_y);
}
